// DGCNN_Propagation_81870666596684
// MI455X (gfx1250) — matrix operand order repaired and checked by reading, not compiled, not run on hardware
//
#include <hip/hip_runtime.h>
#include <hip/hip_bf16.h>

typedef __attribute__((ext_vector_type(16))) _Float16 v16h;
typedef __attribute__((ext_vector_type(8)))  _Float16 v8h;
typedef __attribute__((ext_vector_type(8)))  float    v8f;

#define NEG_SLOPE 0.2f
#define EPS 1e-5f

#if defined(__has_builtin)
#if __has_builtin(__builtin_amdgcn_global_load_async_to_lds_b128) && \
    __has_builtin(__builtin_amdgcn_s_wait_asynccnt)
#define USE_ASYNC_LDS 1
#endif
#endif
#ifndef USE_ASYNC_LDS
#define USE_ASYNC_LDS 0
#endif

// Exact parameter types demanded by the async-LDS builtin (from hipcc
// diagnostics): int __attribute__((vector_size(16))) * in addrspace(1)/(3).
typedef int vec4i __attribute__((vector_size(16)));
typedef __attribute__((address_space(1))) vec4i as1_vec4i;
typedef __attribute__((address_space(3))) vec4i as3_vec4i;

// ---------------------------------------------------------------------------
// Transpose [B, C, G] f32 -> [B, G, C] f16 (tiled, LDS)
// ---------------------------------------------------------------------------
__global__ __launch_bounds__(256) void transpose_f32_to_f16_kernel(
    const float* __restrict__ src, _Float16* __restrict__ dst, int C, int G) {
  __shared__ float tile[32][33];
  int b  = blockIdx.z;
  int g0 = blockIdx.x * 32;
  int c0 = blockIdx.y * 32;
  for (int i = threadIdx.y; i < 32; i += blockDim.y) {
    tile[i][threadIdx.x] = src[((size_t)b * C + (c0 + i)) * G + g0 + threadIdx.x];
  }
  __syncthreads();
  for (int i = threadIdx.y; i < 32; i += blockDim.y) {
    dst[((size_t)b * G + (g0 + i)) * C + c0 + threadIdx.x] =
        (_Float16)tile[threadIdx.x][i];
  }
}

// ---------------------------------------------------------------------------
// f32 -> f16 elementwise (weights)
// ---------------------------------------------------------------------------
__global__ void f32_to_f16_kernel(const float* __restrict__ src,
                                  _Float16* __restrict__ dst, int n) {
  int i = blockIdx.x * blockDim.x + threadIdx.x;
  if (i < n) dst[i] = (_Float16)src[i];
}

// ---------------------------------------------------------------------------
// kNN top-4 (squared distance). One thread per query; candidates staged in LDS.
// cq: [B,3,Gq], ck: [B,3,Gk], idx: [B,Gq,4]
// ---------------------------------------------------------------------------
__global__ __launch_bounds__(128) void knn4_kernel(
    const float* __restrict__ cq, const float* __restrict__ ck,
    int* __restrict__ idx, int Gq, int Gk) {
  __shared__ float sx[128], sy[128], sz[128];
  int blocksPerB = Gq / 128;
  int b = blockIdx.x / blocksPerB;
  int q = (blockIdx.x % blocksPerB) * 128 + threadIdx.x;

  float qx = cq[((size_t)b * 3 + 0) * Gq + q];
  float qy = cq[((size_t)b * 3 + 1) * Gq + q];
  float qz = cq[((size_t)b * 3 + 2) * Gq + q];

  float bd0 = 1e30f, bd1 = 1e30f, bd2 = 1e30f, bd3 = 1e30f;
  int   bi0 = 0, bi1 = 0, bi2 = 0, bi3 = 0;

  for (int j0 = 0; j0 < Gk; j0 += 128) {
    __syncthreads();
    sx[threadIdx.x] = ck[((size_t)b * 3 + 0) * Gk + j0 + threadIdx.x];
    sy[threadIdx.x] = ck[((size_t)b * 3 + 1) * Gk + j0 + threadIdx.x];
    sz[threadIdx.x] = ck[((size_t)b * 3 + 2) * Gk + j0 + threadIdx.x];
    __syncthreads();
    for (int t = 0; t < 128; ++t) {
      float dx = qx - sx[t], dy = qy - sy[t], dz = qz - sz[t];
      float d = dx * dx + dy * dy + dz * dz;
      int j = j0 + t;
      if (d < bd3) {
        if (d < bd2) {
          bd3 = bd2; bi3 = bi2;
          if (d < bd1) {
            bd2 = bd1; bi2 = bi1;
            if (d < bd0) {
              bd1 = bd0; bi1 = bi0; bd0 = d; bi0 = j;
            } else { bd1 = d; bi1 = j; }
          } else { bd2 = d; bi2 = j; }
        } else { bd3 = d; bi3 = j; }
      }
    }
  }
  int* o = idx + ((size_t)b * Gq + q) * 4;
  o[0] = bi0; o[1] = bi1; o[2] = bi2; o[3] = bi3;
}

// ---------------------------------------------------------------------------
// Edge-feature gather: build e (column-major f16, K = 2*C rows contiguous).
// ft:  [B, Gk, C] f16 (point-major), fqt: [B, Gq, C] f16
// column n = (b*Gq+g)*4+kk; e[n*2C + c] = ft[idx] - fq; e[n*2C + C + c] = fq
// ---------------------------------------------------------------------------
__global__ __launch_bounds__(128) void gather_edge_kernel(
    const _Float16* __restrict__ ft, const _Float16* __restrict__ fqt,
    const int* __restrict__ idx, _Float16* __restrict__ e,
    int C, int Gq, int Gk) {
  int bg = blockIdx.x;            // b*Gq + g
  int b  = bg / Gq;
  const int* id = idx + (size_t)bg * 4;
  int js[4] = {id[0], id[1], id[2], id[3]};
  int rows = 2 * C;
  for (int c = threadIdx.x; c < C; c += 128) {
    float fq = (float)fqt[(size_t)bg * C + c];
    for (int kk = 0; kk < 4; ++kk) {
      float v = (float)ft[((size_t)b * Gk + js[kk]) * C + c];
      _Float16* col = e + ((size_t)bg * 4 + kk) * rows;
      col[c]     = (_Float16)(v - fq);
      col[C + c] = (_Float16)fq;
    }
  }
}

// ---------------------------------------------------------------------------
// WMMA GEMM:  Cmat[M, NTOT] = A[M, KD] (row-major f16) x Bm (col-major f16,
// column n contiguous over KD). Block tile 128x128, 8 waves of 32x64 each,
// K-step 32 via v_wmma_f32_16x16x32_f16. Double-buffered LDS staging; uses
// GLOBAL_LOAD_ASYNC_TO_LDS_B128 (ASYNCcnt) when the toolchain exposes it.
// KD, NTOT compile-time so C-stores fold into immediate offsets.
// ---------------------------------------------------------------------------
template <int KD, int NTOT>
__global__ __launch_bounds__(256) void wmma_gemm_kernel(
    const _Float16* __restrict__ A, const _Float16* __restrict__ Bm,
    float* __restrict__ Cmat) {
  __shared__ __align__(16) _Float16 As[2][128 * 40];  // rows x padded-K
  __shared__ __align__(16) _Float16 Bs[2][128 * 40];  // cols x padded-K

  const int tid  = threadIdx.x;
  const int m0   = blockIdx.y * 128;
  const int n0   = blockIdx.x * 128;
  const int lane = tid & 31;
  const int wave = tid >> 5;
  const int wm   = (wave >> 1) * 32;  // wave row offset in tile
  const int wn   = (wave & 1) * 64;   // wave col offset in tile
  constexpr int NK = KD / 32;

  // Per-thread staging addresses: 2 threads per row/col, 16 halves each
  const int rI  = tid >> 1;
  const int off = (tid & 1) * 16;
  const _Float16* aSrc = A  + (size_t)(m0 + rI) * KD + off;
  const _Float16* bSrc = Bm + (size_t)(n0 + rI) * KD + off;
  const int ldsOff = rI * 40 + off;

  v8f acc[2][4] = {};

  auto stage = [&](int buf, int k0) {
#if USE_ASYNC_LDS
    __builtin_amdgcn_global_load_async_to_lds_b128(
        (as1_vec4i*)(aSrc + k0),     (as3_vec4i*)&As[buf][ldsOff],     0, 0);
    __builtin_amdgcn_global_load_async_to_lds_b128(
        (as1_vec4i*)(aSrc + k0 + 8), (as3_vec4i*)&As[buf][ldsOff + 8], 0, 0);
    __builtin_amdgcn_global_load_async_to_lds_b128(
        (as1_vec4i*)(bSrc + k0),     (as3_vec4i*)&Bs[buf][ldsOff],     0, 0);
    __builtin_amdgcn_global_load_async_to_lds_b128(
        (as1_vec4i*)(bSrc + k0 + 8), (as3_vec4i*)&Bs[buf][ldsOff + 8], 0, 0);
#else
    uint4 a0 = *(const uint4*)(aSrc + k0);
    uint4 a1 = *(const uint4*)(aSrc + k0 + 8);
    uint4 b0 = *(const uint4*)(bSrc + k0);
    uint4 b1 = *(const uint4*)(bSrc + k0 + 8);
    *(uint4*)&As[buf][ldsOff]     = a0;
    *(uint4*)&As[buf][ldsOff + 8] = a1;
    *(uint4*)&Bs[buf][ldsOff]     = b0;
    *(uint4*)&Bs[buf][ldsOff + 8] = b1;
#endif
  };

  auto compute = [&](int buf) {
    // A fragments: lane L<16 row M=L holds K 0..7 & 16..23; lanes 16..31 hold
    // K 8..15 & 24..31 (ISA 7.12.2, 16-bit A 16x32)
    v16h af[2];
    #pragma unroll
    for (int r = 0; r < 2; ++r) {
      int row  = wm + r * 16 + (lane & 15);
      int base = (lane >> 4) * 8;
      v8h lo = *(const v8h*)&As[buf][row * 40 + base];
      v8h hi = *(const v8h*)&As[buf][row * 40 + base + 16];
      af[r] = __builtin_shufflevector(lo, hi, 0, 1, 2, 3, 4, 5, 6, 7,
                                      8, 9, 10, 11, 12, 13, 14, 15);
    }
    // B fragments: lane L<16 = column N=L with K 0..7 & 16..23; lanes
    // 16..31 = column with K 8..15 & 24..31
    v16h bf[4];
    #pragma unroll
    for (int cI = 0; cI < 4; ++cI) {
      int col  = wn + cI * 16 + (lane & 15);
      int koff = (lane >> 4) * 8;
      v8h lo = *(const v8h*)&Bs[buf][col * 40 + koff];
      v8h hi = *(const v8h*)&Bs[buf][col * 40 + koff + 16];
      bf[cI] = __builtin_shufflevector(lo, hi, 0, 1, 2, 3, 4, 5, 6, 7,
                                       8, 9, 10, 11, 12, 13, 14, 15);
    }
    #pragma unroll
    for (int r = 0; r < 2; ++r)
      #pragma unroll
      for (int cI = 0; cI < 4; ++cI)
        acc[r][cI] = __builtin_amdgcn_wmma_f32_16x16x32_f16(
            false, af[r], false, bf[cI], (short)0, acc[r][cI], false, false);
  };

#if USE_ASYNC_LDS
  stage(0, 0);
  #pragma unroll 2
  for (int kt = 0; kt < NK; ++kt) {
    int buf = kt & 1;
    __builtin_amdgcn_s_wait_asynccnt(0);   // my tile-kt loads landed in LDS
    __syncthreads();                       // everyone's landed; prev compute done
    if (kt + 1 < NK) stage(buf ^ 1, (kt + 1) * 32);  // prefetch next tile
    compute(buf);
  }
#else
  #pragma unroll 2
  for (int kt = 0; kt < NK; ++kt) {
    int buf = kt & 1;
    stage(buf, kt * 32);
    __syncthreads();
    compute(buf);
  }
#endif

  // Store: VGPR j -> M = j + 8*(lane>>4), N = lane&15 (ISA 7.12.2 f32 C/D).
  // NTOT compile-time => row strides become instruction-immediate offsets.
  {
    float* base = Cmat + (size_t)(m0 + wm + ((lane >> 4) << 3)) * NTOT +
                  (n0 + wn + (lane & 15));
    #pragma unroll
    for (int r = 0; r < 2; ++r)
      #pragma unroll
      for (int cI = 0; cI < 4; ++cI)
        #pragma unroll
        for (int j = 0; j < 8; ++j)
          base[(size_t)(r * 16 + j) * NTOT + cI * 16] = acc[r][cI][j];
  }
}

// ---------------------------------------------------------------------------
// GroupNorm stats: partial sum / sumsq per (batch, group) via atomics.
// y: [M, Ntot] f32, stats: [B*4][2] f32. colsPerBatch = Gq*4.
// ---------------------------------------------------------------------------
__global__ __launch_bounds__(256) void gn_stats_kernel(
    const float* __restrict__ y, float* __restrict__ stats,
    int M, int Ntot, int colsPerBatch) {
  __shared__ float s1[256], s2[256];
  int tid = threadIdx.x;
  int o   = blockIdx.y;
  int col = blockIdx.x * 256 + tid;
  float v = y[(size_t)o * Ntot + col];
  s1[tid] = v;
  s2[tid] = v * v;
  __syncthreads();
  for (int srd = 128; srd > 0; srd >>= 1) {
    if (tid < srd) { s1[tid] += s1[tid + srd]; s2[tid] += s2[tid + srd]; }
    __syncthreads();
  }
  if (tid == 0) {
    int b  = (blockIdx.x * 256) / colsPerBatch;
    int gi = o / (M / 4);
    atomicAdd(&stats[(b * 4 + gi) * 2 + 0], s1[0]);
    atomicAdd(&stats[(b * 4 + gi) * 2 + 1], s2[0]);
  }
}

__global__ void zero_kernel(float* p, int n) {
  int i = blockIdx.x * blockDim.x + threadIdx.x;
  if (i < n) p[i] = 0.0f;
}

// ---------------------------------------------------------------------------
// Fused normalize + affine + LeakyReLU + max over k. out: [B, M, Gq] f32
// ---------------------------------------------------------------------------
__global__ __launch_bounds__(256) void gn_apply_max_kernel(
    const float* __restrict__ y, const float* __restrict__ stats,
    const float* __restrict__ gamma, const float* __restrict__ beta,
    float* __restrict__ out, int M, int Gq, int Ntot, float invCnt, int total) {
  int i = blockIdx.x * blockDim.x + threadIdx.x;
  if (i >= total) return;
  int g = i % Gq;
  int o = (i / Gq) % M;
  int b = i / (Gq * M);
  int gi = o / (M / 4);
  float s  = stats[(b * 4 + gi) * 2 + 0];
  float ss = stats[(b * 4 + gi) * 2 + 1];
  float mean = s * invCnt;
  float var  = ss * invCnt - mean * mean;
  float inv  = rsqrtf(var + EPS);
  float ga = gamma[o], be = beta[o];
  size_t base = (size_t)o * Ntot + ((size_t)b * Gq + g) * 4;
  float m = -1e30f;
  #pragma unroll
  for (int kk = 0; kk < 4; ++kk) {
    float v = (y[base + kk] - mean) * inv * ga + be;
    v = (v > 0.0f) ? v : NEG_SLOPE * v;
    m = fmaxf(m, v);
  }
  out[((size_t)b * M + o) * Gq + g] = m;
}

// ---------------------------------------------------------------------------
// Host orchestration
// ---------------------------------------------------------------------------
extern "C" void kernel_launch(void* const* d_in, const int* in_sizes, int n_in,
                              void* d_out, int out_size, void* d_ws, size_t ws_size,
                              hipStream_t stream) {
  constexpr int B = 16, C = 384, Gs = 4096, Gd = 1024;
  constexpr int M1 = 512, K1 = 2 * C;     // 768
  constexpr int M2 = C,   K2 = 2 * M1;    // 384, 1024
  constexpr int Ntot = B * Gd * 4;        // 65536

  const float* coor   = (const float*)d_in[0];
  const float* f      = (const float*)d_in[1];
  const float* coor_q = (const float*)d_in[2];
  const float* f_q    = (const float*)d_in[3];
  const float* W1     = (const float*)d_in[4];
  const float* g1     = (const float*)d_in[5];
  const float* b1     = (const float*)d_in[6];
  const float* W2     = (const float*)d_in[7];
  const float* g2     = (const float*)d_in[8];
  const float* b2     = (const float*)d_in[9];
  float* out = (float*)d_out;

  // Static workspace partition
  char*  w   = (char*)d_ws;
  size_t off = 0;
  auto alloc = [&](size_t bytes) -> void* {
    void* p = w + off;
    off = (off + bytes + 255) & ~(size_t)255;
    return p;
  };
  int*      idx1  = (int*)alloc((size_t)B * Gd * 4 * sizeof(int));
  int*      idx2  = (int*)alloc((size_t)B * Gd * 4 * sizeof(int));
  _Float16* ft    = (_Float16*)alloc((size_t)B * Gs * C  * 2);
  _Float16* fqt   = (_Float16*)alloc((size_t)B * Gd * C  * 2);
  _Float16* ht    = (_Float16*)alloc((size_t)B * Gd * M1 * 2);
  _Float16* w1h   = (_Float16*)alloc((size_t)M1 * K1 * 2);
  _Float16* w2h   = (_Float16*)alloc((size_t)M2 * K2 * 2);
  _Float16* e     = (_Float16*)alloc((size_t)K2 * Ntot * 2);  // max(K1,K2) rows
  float*    y     = (float*)alloc((size_t)M1 * Ntot * sizeof(float));
  float*    h     = (float*)alloc((size_t)B * M1 * Gd * sizeof(float));
  float*    stats = (float*)alloc(128 * sizeof(float));

  dim3 t256(256), t128(128);

  // --- Stage 1 -------------------------------------------------------------
  knn4_kernel<<<dim3(B * Gd / 128), t128, 0, stream>>>(coor_q, coor, idx1, Gd, Gs);
  transpose_f32_to_f16_kernel<<<dim3(Gs / 32, C / 32, B), dim3(32, 8), 0, stream>>>(f, ft, C, Gs);
  transpose_f32_to_f16_kernel<<<dim3(Gd / 32, C / 32, B), dim3(32, 8), 0, stream>>>(f_q, fqt, C, Gd);
  f32_to_f16_kernel<<<dim3((M1 * K1 + 255) / 256), t256, 0, stream>>>(W1, w1h, M1 * K1);
  f32_to_f16_kernel<<<dim3((M2 * K2 + 255) / 256), t256, 0, stream>>>(W2, w2h, M2 * K2);
  gather_edge_kernel<<<dim3(B * Gd), t128, 0, stream>>>(ft, fqt, idx1, e, C, Gd, Gs);
  // GEMM1: [512 x 768] x [768 x 65536]
  wmma_gemm_kernel<K1, Ntot><<<dim3(Ntot / 128, M1 / 128), t256, 0, stream>>>(w1h, e, y);
  zero_kernel<<<dim3(1), t128, 0, stream>>>(stats, 128);
  gn_stats_kernel<<<dim3(Ntot / 256, M1), t256, 0, stream>>>(y, stats, M1, Ntot, Gd * 4);
  {
    int total = B * M1 * Gd;
    float invCnt = 1.0f / ((float)(M1 / 4) * Gd * 4);
    gn_apply_max_kernel<<<dim3((total + 255) / 256), t256, 0, stream>>>(
        y, stats, g1, b1, h, M1, Gd, Ntot, invCnt, total);
  }

  // --- Stage 2 -------------------------------------------------------------
  knn4_kernel<<<dim3(B * Gd / 128), t128, 0, stream>>>(coor_q, coor_q, idx2, Gd, Gd);
  transpose_f32_to_f16_kernel<<<dim3(Gd / 32, M1 / 32, B), dim3(32, 8), 0, stream>>>(h, ht, M1, Gd);
  gather_edge_kernel<<<dim3(B * Gd), t128, 0, stream>>>(ht, ht, idx2, e, M1, Gd, Gd);
  // GEMM2: [384 x 1024] x [1024 x 65536]
  wmma_gemm_kernel<K2, Ntot><<<dim3(Ntot / 128, M2 / 128), t256, 0, stream>>>(w2h, e, y);
  zero_kernel<<<dim3(1), t128, 0, stream>>>(stats, 128);
  gn_stats_kernel<<<dim3(Ntot / 256, M2), t256, 0, stream>>>(y, stats, M2, Ntot, Gd * 4);
  {
    int total = B * M2 * Gd;
    float invCnt = 1.0f / ((float)(M2 / 4) * Gd * 4);
    gn_apply_max_kernel<<<dim3((total + 255) / 256), t256, 0, stream>>>(
        y, stats, g2, b2, out, M2, Gd, Ntot, invCnt, total);
  }
}
